// SetContrastive_43894565765616
// MI455X (gfx1250) — matrix operand order repaired and checked by reading, not compiled, not run on hardware
//
#include <hip/hip_runtime.h>
#include <hip/hip_bf16.h>
#include <math.h>

typedef _Float16 v16h __attribute__((ext_vector_type(16)));
typedef _Float16 v8h  __attribute__((ext_vector_type(8)));
typedef _Float16 v4h  __attribute__((ext_vector_type(4)));
typedef float    v8f  __attribute__((ext_vector_type(8)));
typedef float    v4f  __attribute__((ext_vector_type(4)));

#define HDIM 128
#define BSZ  64
#define SQ   32
#define SD   256

// ---------------------------------------------------------------------------
// Kernel 1: per-token L2 normalize, split into f16 hi/lo parts (Markidis),
// fold 0/1 mask into the stored operands, and accumulate the per-batch
// sum-vector + trace for the off-diagonal-mean term.
// One block per batch element; 8 waves, each wave normalizes tokens s = w+8k.
// ---------------------------------------------------------------------------
__global__ void norm_split_kernel(const float* __restrict__ src,
                                  const int*   __restrict__ mask,
                                  _Float16* __restrict__ outh,
                                  _Float16* __restrict__ outl,
                                  float*    __restrict__ avg_out,
                                  int S, float inv_div) {
  __shared__ float sumvec[HDIM];
  __shared__ float trace;
  const int b    = blockIdx.x;
  const int t    = threadIdx.x;
  const int w    = t >> 5;
  const int lane = t & 31;

  if (t < HDIM) sumvec[t] = 0.0f;
  if (t == 0)   trace = 0.0f;
  __syncthreads();

  const float* base = src + (size_t)b * S * HDIM;

  for (int s = w; s < S; s += 8) {
    v4f x = *(const v4f*)(base + (size_t)s * HDIM + lane * 4);
    float ss = x.x * x.x + x.y * x.y + x.z * x.z + x.w * x.w;
    #pragma unroll
    for (int off = 16; off; off >>= 1) ss += __shfl_xor(ss, off, 32);
    const float nrm = sqrtf(ss);
    const float m   = fmaxf(nrm, 1e-12f);   // matches F.normalize eps
    const float mk  = (float)mask[b * S + s];

    float y[4] = { x.x / m, x.y / m, x.z / m, x.w / m };
    v4h vh, vl;
    #pragma unroll
    for (int i = 0; i < 4; i++) {
      const float yi = y[i];
      const _Float16 hi = (_Float16)yi;
      const _Float16 lo = (_Float16)(yi - (float)hi);
      vh[i] = (mk != 0.0f) ? hi : (_Float16)0.0f;
      vl[i] = (mk != 0.0f) ? lo : (_Float16)0.0f;
      atomicAdd(&sumvec[lane * 4 + i], yi);       // unmasked q-hat for avg_sim
    }
    const size_t row = ((size_t)b * S + s) * HDIM + lane * 4;
    *(v4h*)(outh + row) = vh;
    *(v4h*)(outl + row) = vl;
    if (lane == 0) atomicAdd(&trace, ss / (m * m));   // == ||q_hat||^2
  }
  __syncthreads();

  if (w == 0) {
    float s2 = 0.0f;
    #pragma unroll
    for (int i = 0; i < 4; i++) {
      const float v = sumvec[lane * 4 + i];
      s2 += v * v;
    }
    #pragma unroll
    for (int off = 16; off; off >>= 1) s2 += __shfl_xor(s2, off, 32);
    if (lane == 0) avg_out[b] = (s2 - trace) * inv_div;
  }
}

// ---------------------------------------------------------------------------
// Kernel 2: scores[a][b] = max over (s,t) of (mask-folded) q_hat[a,s]·d_hat[b,t].
// Block (b,a): the 32x256 product region = 2x16 tiles of 16x16; 8 waves x 4
// tiles. K=128 handled as 4 chunks of 32 with split-f16 WMMA:
//   C += Ah*Bh + Ah*Bl + Al*Bh   (f32 accumulate, ~fp32 accuracy)
// Fragments loaded straight from global in the native WMMA VGPR layout.
// ---------------------------------------------------------------------------
__global__ void maxmax_scores_kernel(const _Float16* __restrict__ qh,
                                     const _Float16* __restrict__ ql,
                                     const _Float16* __restrict__ dh,
                                     const _Float16* __restrict__ dl,
                                     float* __restrict__ scores) {
  __shared__ float smax[8];
  const int b    = blockIdx.x;   // document batch index
  const int a    = blockIdx.y;   // query batch index
  const int t    = threadIdx.x;
  const int w    = t >> 5;
  const int lane = t & 31;
  const int grp    = lane >> 4;  // 0: lanes 0-15, 1: lanes 16-31
  const int lane16 = lane & 15;

  float tmax = -INFINITY;

  #pragma unroll
  for (int i = 0; i < 4; i++) {
    const int tid = w * 4 + i;           // 0..31
    const int mt  = tid >> 4;            // 0..1   (q row tile)
    const int nt  = tid & 15;            // 0..15  (d col tile)

    const size_t arow = (size_t)a * SQ + mt * 16 + lane16;  // A: row = lane%16
    const size_t ncol = (size_t)b * SD + nt * 16 + lane16;  // B: col = lane%16
    const _Float16* qhp = qh + arow * HDIM;
    const _Float16* qlp = ql + arow * HDIM;
    const _Float16* dhp = dh + ncol * HDIM;
    const _Float16* dlp = dl + ncol * HDIM;

    v8f c = {};
    #pragma unroll
    for (int kc = 0; kc < 4; kc++) {
      // A 16x32 f16 layout: lane holds K = base..base+7 and base+16..base+23,
      // base = grp*8 (ISA 7.12.2, 16-bit A-matrix).
      const int ab = kc * 32 + grp * 8;
      v8h a0 = *(const v8h*)(qhp + ab);
      v8h a1 = *(const v8h*)(qhp + ab + 16);
      v8h l0 = *(const v8h*)(qlp + ab);
      v8h l1 = *(const v8h*)(qlp + ab + 16);
      v16h ah, al;
      #pragma unroll
      for (int j = 0; j < 8; j++) {
        ah[j] = a0[j]; ah[8 + j] = a1[j];
        al[j] = l0[j]; al[8 + j] = l1[j];
      }
      // B 32x16 f16 layout: lane holds K = base..base+7 and base+16..base+23 (same as A), col = lane%16.
      const int bb = kc * 32 + grp * 8;
      v8h b0 = *(const v8h*)(dhp + bb);
      v8h b1 = *(const v8h*)(dhp + bb + 16);
      v8h m0 = *(const v8h*)(dlp + bb);
      v8h m1 = *(const v8h*)(dlp + bb + 16);
      v16h bh, bl;
      #pragma unroll
      for (int j = 0; j < 8; j++) {
        bh[j] = b0[j]; bh[8 + j] = b1[j];
        bl[j] = m0[j]; bl[8 + j] = m1[j];
      }

      c = __builtin_amdgcn_wmma_f32_16x16x32_f16(false, ah, false, bh,
                                                 (short)0, c, false, false);
      c = __builtin_amdgcn_wmma_f32_16x16x32_f16(false, ah, false, bl,
                                                 (short)0, c, false, false);
      c = __builtin_amdgcn_wmma_f32_16x16x32_f16(false, al, false, bh,
                                                 (short)0, c, false, false);
    }
    #pragma unroll
    for (int j = 0; j < 8; j++) tmax = fmaxf(tmax, c[j]);
  }

  #pragma unroll
  for (int off = 16; off; off >>= 1) tmax = fmaxf(tmax, __shfl_xor(tmax, off, 32));
  if (lane == 0) smax[w] = tmax;
  __syncthreads();
  if (t == 0) {
    float m = smax[0];
    #pragma unroll
    for (int i = 1; i < 8; i++) m = fmaxf(m, smax[i]);
    scores[a * BSZ + b] = m;
  }
}

// ---------------------------------------------------------------------------
// Kernel 3: loss = mean_a( logsumexp_b scores[a,b] - scores[a,a] )
//                + mean(avgq) + mean(avgd).       (TEMPERATURE = DIV_COEFF = 1)
// ---------------------------------------------------------------------------
__global__ void loss_kernel(const float* __restrict__ scores,
                            const float* __restrict__ avgq,
                            const float* __restrict__ avgd,
                            float* __restrict__ out) {
  __shared__ float red[2];
  const int a = threadIdx.x;          // 0..63, 2 waves
  const float* row = scores + a * BSZ;
  float m = -INFINITY;
  for (int j = 0; j < BSZ; j++) m = fmaxf(m, row[j]);
  float s = 0.0f;
  for (int j = 0; j < BSZ; j++) s += expf(row[j] - m);
  const float lse = m + logf(s);
  float v = lse - row[a] + avgq[a] + avgd[a];
  #pragma unroll
  for (int off = 16; off; off >>= 1) v += __shfl_xor(v, off, 32);
  const int w = a >> 5, lane = a & 31;
  if (lane == 0) red[w] = v;
  __syncthreads();
  if (a == 0) out[0] = (red[0] + red[1]) * (1.0f / BSZ);
}

// ---------------------------------------------------------------------------
extern "C" void kernel_launch(void* const* d_in, const int* in_sizes, int n_in,
                              void* d_out, int out_size, void* d_ws, size_t ws_size,
                              hipStream_t stream) {
  const float* q_emb  = (const float*)d_in[0];   // [64,32,128]  f32
  const float* d_emb  = (const float*)d_in[1];   // [64,256,128] f32
  const int*   q_mask = (const int*)d_in[2];     // [64,32]      i32
  const int*   d_mask = (const int*)d_in[3];     // [64,256]     i32

  const size_t qn = (size_t)BSZ * SQ * HDIM;   // 262144 halves
  const size_t dn = (size_t)BSZ * SD * HDIM;   // 2097152 halves

  _Float16* qh = (_Float16*)d_ws;
  _Float16* ql = qh + qn;
  _Float16* dh = ql + qn;
  _Float16* dl = dh + dn;
  float* scores = (float*)(dl + dn);           // 64*64 f32, 256B-aligned offset
  float* avgq   = scores + BSZ * BSZ;
  float* avgd   = avgq + BSZ;
  (void)in_sizes; (void)n_in; (void)out_size; (void)ws_size;

  norm_split_kernel<<<BSZ, 256, 0, stream>>>(q_emb, q_mask, qh, ql, avgq,
                                             SQ, 1.0f / (float)(SQ * (SQ - 1)));
  norm_split_kernel<<<BSZ, 256, 0, stream>>>(d_emb, d_mask, dh, dl, avgd,
                                             SD, 1.0f / (float)(SD * (SD - 1)));
  maxmax_scores_kernel<<<dim3(BSZ, BSZ), 256, 0, stream>>>(qh, ql, dh, dl, scores);
  loss_kernel<<<1, BSZ, 0, stream>>>(scores, avgq, avgd, (float*)d_out);
}
